// SNNAlif_45792941310312
// MI455X (gfx1250) — matrix operand order repaired and checked by reading, not compiled, not run on hardware
//
#include <hip/hip_runtime.h>
#include <hip/hip_fp16.h>

typedef __attribute__((ext_vector_type(16))) _Float16 v16h;
typedef __attribute__((ext_vector_type(8)))  float    v8f;

// ---- problem sizes ----
#define T_STEPS 500
#define B_SZ    64
#define F_SZ    512
#define H_SZ    1024
#define O_SZ    128

// ---- LSNN constants ----
#define DT            1e-3f
#define TAU_SYN_INV   200.0f
#define TAU_MEM_INV   100.0f
#define TAU_ADAPT_INV (1.0f / 700.0f)
#define V_LEAK        0.0f
#define V_TH          1.0f
#define BETA          1.8f
#define FILT_A        (DT * 223.1435511314f)

// =====================================================================
// Generic WMMA GEMM:  C[M,N] = A[M,K] * Bw[N,K]^T (+ bias[N])
// A is f32 or f16 (converted to f16 in LDS); Bw is f32 weights.
// Macro-tile 64(M) x 128(N), 256 threads = 8 waves, each wave 32x32.
// =====================================================================
template <typename AT, bool HAS_BIAS>
__global__ __launch_bounds__(256) void wmma_gemm_kernel(
    const AT* __restrict__ A, const float* __restrict__ Bw,
    const float* __restrict__ bias, float* __restrict__ C,
    int M, int N, int K) {
  constexpr int KC  = 32;
  constexpr int LDA = KC + 2;                 // pad to avoid bank conflicts
  __shared__ _Float16 As[64  * LDA];
  __shared__ _Float16 Bs[128 * LDA];

  const int tid  = threadIdx.x;
  const int lane = tid & 31;
  const int wave = tid >> 5;
  const int g    = lane >> 4;                  // lane group 0/1
  const int lr   = lane & 15;
  const int mBlk = blockIdx.x * 64;
  const int nBlk = blockIdx.y * 128;
  const int mw   = (wave & 1) * 32;            // wave offset in macro-tile
  const int nw   = (wave >> 1) * 32;

  v8f acc[2][2];
  #pragma unroll
  for (int i = 0; i < 2; ++i)
    #pragma unroll
    for (int j = 0; j < 2; ++j)
      #pragma unroll
      for (int r = 0; r < 8; ++r) acc[i][j][r] = 0.0f;

  for (int kb = 0; kb < K; kb += KC) {
    __syncthreads();
    {   // stage A tile 64x32 (8 contiguous elems / thread), convert -> f16
      int r = tid >> 2;
      int c = (tid & 3) * 8;
      const AT* src = A + (size_t)(mBlk + r) * K + kb + c;
      _Float16* dst = As + r * LDA + c;
      #pragma unroll
      for (int i = 0; i < 8; ++i) dst[i] = (_Float16)src[i];
    }
    {   // stage B tile 128x32
      #pragma unroll
      for (int p = 0; p < 2; ++p) {
        int idx = tid + p * 256;
        int r = idx >> 2;
        int c = (idx & 3) * 8;
        const float* src = Bw + (size_t)(nBlk + r) * K + kb + c;
        _Float16* dst = Bs + r * LDA + c;
        #pragma unroll
        for (int i = 0; i < 8; ++i) dst[i] = (_Float16)src[i];
      }
    }
    __syncthreads();

    union { v16h v; unsigned u[8]; } fa[2], fb[2];
    #pragma unroll
    for (int s = 0; s < 2; ++s) {
      // A fragment: 16x32 f16 layout — K contiguous, lanes 16-31 hold K=16..31
      const _Float16* row = As + (mw + s * 16 + lr) * LDA;
      #pragma unroll
      for (int vv = 0; vv < 8; ++vv) {
        int kk = (g << 4) + (vv << 1);
        fa[s].u[vv] = *(const unsigned*)(row + kk);
      }
      // B fragment: 32x16 f16 layout — K contiguous, lanes 16-31 hold K=16..31
      const _Float16* rowb = Bs + (nw + s * 16 + lr) * LDA + (g << 4);
      #pragma unroll
      for (int vv = 0; vv < 8; ++vv)
        fb[s].u[vv] = *(const unsigned*)(rowb + (vv << 1));
    }
    #pragma unroll
    for (int i = 0; i < 2; ++i)
      #pragma unroll
      for (int j = 0; j < 2; ++j)
        acc[i][j] = __builtin_amdgcn_wmma_f32_16x16x32_f16(
            false, fa[i].v, false, fb[j].v, (short)0, acc[i][j], false, false);
  }

  // epilogue: C layout VGPR r, lanes0-15 -> M=r, lanes16-31 -> M=8+r; N=lane%16
  #pragma unroll
  for (int i = 0; i < 2; ++i)
    #pragma unroll
    for (int j = 0; j < 2; ++j) {
      int n = nBlk + nw + j * 16 + lr;
      float bv = 0.0f;
      if constexpr (HAS_BIAS) bv = bias[n];
      #pragma unroll
      for (int r = 0; r < 8; ++r) {
        int m = mBlk + mw + i * 16 + r + g * 8;
        C[(size_t)m * N + n] = acc[i][j][r] + bv;
      }
    }
}

// =====================================================================
// Grid-wide barrier (release/acquire over 16 resident blocks)
// =====================================================================
__device__ inline void gridBarrier(unsigned* cnt, unsigned* gen, unsigned nb) {
  __syncthreads();
  if (threadIdx.x == 0) {
    __threadfence();
    unsigned old  = __hip_atomic_load(gen, __ATOMIC_RELAXED, __HIP_MEMORY_SCOPE_AGENT);
    unsigned prev = __hip_atomic_fetch_add(cnt, 1u, __ATOMIC_ACQ_REL, __HIP_MEMORY_SCOPE_AGENT);
    if (prev == nb - 1u) {
      __hip_atomic_store(cnt, 0u, __ATOMIC_RELAXED, __HIP_MEMORY_SCOPE_AGENT);
      __hip_atomic_fetch_add(gen, 1u, __ATOMIC_ACQ_REL, __HIP_MEMORY_SCOPE_AGENT);
    } else {
      while (__hip_atomic_load(gen, __ATOMIC_ACQUIRE, __HIP_MEMORY_SCOPE_AGENT) == old)
        __builtin_amdgcn_s_sleep(2);
    }
  }
  __syncthreads();
}

// =====================================================================
// Persistent ALIF scan. 16 WGs x 256 threads; WG g owns H columns
// [g*64, g*64+64). Its f16 w_rec slice (64x1024 ~ 128KB) lives in the
// 320KB WGP LDS for all 500 steps; z streams from L2 (128KB/step).
// =====================================================================
__global__ __launch_bounds__(256) void lsnn_scan_kernel(
    const float* __restrict__ xin, const float* __restrict__ w_rec,
    _Float16* __restrict__ spikes, float* __restrict__ v_s,
    float* __restrict__ i_s, float* __restrict__ b_s,
    unsigned* __restrict__ bar) {
  extern __shared__ _Float16 wl[];             // [64][H_SZ + 2]
  constexpr int LDW = H_SZ + 2;

  const int tid  = threadIdx.x;
  const int lane = tid & 31;
  const int wave = tid >> 5;
  const int g    = lane >> 4;
  const int lr   = lane & 15;
  const int ng0  = blockIdx.x * 64;

  // preload w_rec slice, converted to f16, [n][k] row-major (coalesced in k)
  for (int idx = tid; idx < 64 * H_SZ; idx += 256) {
    int n = idx >> 10;
    int k = idx & (H_SZ - 1);
    wl[n * LDW + k] = (_Float16)w_rec[(size_t)(ng0 + n) * H_SZ + k];
  }
  __syncthreads();

  // wave -> one 16-row M slice, two 16-col N tiles (16 tiles / 8 waves)
  const int mi    = (wave & 3) * 16;
  const int nbase = (wave >> 2) * 32;

  for (int t = 0; t < T_STEPS; ++t) {
    v8f acc[2];
    #pragma unroll
    for (int j = 0; j < 2; ++j)
      #pragma unroll
      for (int r = 0; r < 8; ++r) acc[j][r] = 0.0f;

    if (t > 0) {
      const _Float16* zprev = spikes + (size_t)(t - 1) * B_SZ * H_SZ;
      const _Float16* arow  = zprev + (size_t)(mi + lr) * H_SZ;
      for (int kb = 0; kb < H_SZ; kb += 32) {
        union { v16h v; unsigned u[8]; } fa, fb0, fb1;
        #pragma unroll
        for (int vv = 0; vv < 8; ++vv) {
          int kk = kb + (g << 4) + (vv << 1);
          fa.u[vv] = *(const unsigned*)(arow + kk);
        }
        const _Float16* rb0 = wl + (nbase + lr) * LDW + kb + (g << 4);
        const _Float16* rb1 = wl + (nbase + 16 + lr) * LDW + kb + (g << 4);
        #pragma unroll
        for (int vv = 0; vv < 8; ++vv) {
          fb0.u[vv] = *(const unsigned*)(rb0 + (vv << 1));
          fb1.u[vv] = *(const unsigned*)(rb1 + (vv << 1));
        }
        acc[0] = __builtin_amdgcn_wmma_f32_16x16x32_f16(
            false, fa.v, false, fb0.v, (short)0, acc[0], false, false);
        acc[1] = __builtin_amdgcn_wmma_f32_16x16x32_f16(
            false, fa.v, false, fb1.v, (short)0, acc[1], false, false);
      }
    }

    // fused ALIF state update on this wave's C elements (exclusive owner)
    const float* xt = xin + (size_t)t * B_SZ * H_SZ;
    _Float16*    zt = spikes + (size_t)t * B_SZ * H_SZ;
    #pragma unroll
    for (int j = 0; j < 2; ++j) {
      int h = ng0 + nbase + j * 16 + lr;
      #pragma unroll
      for (int r = 0; r < 8; ++r) {
        int m = mi + r + g * 8;
        size_t sidx = (size_t)m * H_SZ + h;
        float i_jump = i_s[sidx] + xt[sidx] + acc[j][r];
        float i_new  = i_jump - (DT * TAU_SYN_INV) * i_jump;
        float v0     = v_s[sidx];
        float v_dec  = v0 + (DT * TAU_MEM_INV) * ((V_LEAK - v0) + i_jump);
        float b0     = b_s[sidx];
        float b_dec  = b0 + (DT * TAU_ADAPT_INV) * (V_TH - b0);
        float z      = (v_dec - b_dec) > 0.0f ? 1.0f : 0.0f;
        i_s[sidx] = i_new;
        v_s[sidx] = (1.0f - z) * v_dec;
        b_s[sidx] = b_dec + z * (TAU_ADAPT_INV * DT * BETA);
        zt[sidx]  = (_Float16)z;
      }
    }
    gridBarrier(bar, bar + 1, gridDim.x);
  }
}

// ---- exp filter: 8192 independent chains over T ----
__global__ void lsnn_filter_kernel(const float* __restrict__ lin,
                                   float* __restrict__ out) {
  int id = blockIdx.x * blockDim.x + threadIdx.x;
  if (id >= B_SZ * O_SZ) return;
  float y = 0.0f;
  for (int t = 0; t < T_STEPS; ++t) {
    float l = lin[(size_t)t * B_SZ * O_SZ + id];
    y = y + FILT_A * (l - y);
    out[(size_t)t * B_SZ * O_SZ + id] = y;
  }
}

__global__ void lsnn_zero_kernel(float* p, int n) {
  int i = blockIdx.x * blockDim.x + threadIdx.x;
  if (i < n) p[i] = 0.0f;
}

// =====================================================================
extern "C" void kernel_launch(void* const* d_in, const int* in_sizes, int n_in,
                              void* d_out, int out_size, void* d_ws, size_t ws_size,
                              hipStream_t stream) {
  (void)in_sizes; (void)n_in; (void)out_size; (void)ws_size;
  const float* x     = (const float*)d_in[0];   // [T,B,F]
  const float* w_in  = (const float*)d_in[1];   // [H,F]
  const float* w_rec = (const float*)d_in[2];   // [H,H]
  const float* w_out = (const float*)d_in[3];   // [O,H]
  const float* b_out = (const float*)d_in[4];   // [O]
  float* out = (float*)d_out;                   // [T,B,O]

  char* ws = (char*)d_ws;
  float*    xin    = (float*)ws;     ws += (size_t)T_STEPS * B_SZ * H_SZ * 4;
  _Float16* spikes = (_Float16*)ws;  ws += (size_t)T_STEPS * B_SZ * H_SZ * 2;
  float*    lin    = (float*)ws;     ws += (size_t)T_STEPS * B_SZ * O_SZ * 4;
  float*    v_s    = (float*)ws;
  float*    i_s    = v_s + (size_t)B_SZ * H_SZ;
  float*    b_s    = i_s + (size_t)B_SZ * H_SZ;
  unsigned* bar    = (unsigned*)(b_s + (size_t)B_SZ * H_SZ);

  // zero states + barrier words every launch (graph-replay safe)
  int nz = 3 * B_SZ * H_SZ + 2;
  lsnn_zero_kernel<<<(nz + 255) / 256, 256, 0, stream>>>(v_s, nz);

  // 1) xin = x @ w_in^T   (M=32000, N=1024, K=512)
  wmma_gemm_kernel<float, false><<<dim3(500, 8), 256, 0, stream>>>(
      x, w_in, nullptr, xin, T_STEPS * B_SZ, H_SZ, F_SZ);

  // 2) persistent ALIF scan, 16 WGs, 128KB+pad dynamic LDS per WG
  int ldsBytes = 64 * (H_SZ + 2) * (int)sizeof(_Float16);
  (void)hipFuncSetAttribute(reinterpret_cast<const void*>(lsnn_scan_kernel),
                            hipFuncAttributeMaxDynamicSharedMemorySize, ldsBytes);
  lsnn_scan_kernel<<<16, 256, ldsBytes, stream>>>(xin, w_rec, spikes,
                                                  v_s, i_s, b_s, bar);

  // 3) lin = spikes @ w_out^T + b_out   (M=32000, N=128, K=1024)
  wmma_gemm_kernel<_Float16, true><<<dim3(500, 1), 256, 0, stream>>>(
      spikes, w_out, b_out, lin, T_STEPS * B_SZ, O_SZ, H_SZ);

  // 4) exponential filter
  lsnn_filter_kernel<<<(B_SZ * O_SZ + 255) / 256, 256, 0, stream>>>(lin, out);
}
